// LSTM_42949672961652
// MI455X (gfx1250) — matrix operand order repaired and checked by reading, not compiled, not run on hardware
//
#include <hip/hip_runtime.h>
#include <hip/hip_bf16.h>

typedef __attribute__((ext_vector_type(16))) _Float16     v16h;
typedef __attribute__((ext_vector_type(8)))  float        v8f;
typedef __attribute__((ext_vector_type(4)))  unsigned int v4u;

union Frag16 { v16h h; v4u u[2]; };

#define T_DIM 512
#define B_DIM 64
#define D_DIM 512
#define H_DIM 512
#define L_DIM 2
#define G_DIM 2048   // 4*H
#define TB_DIM (T_DIM * B_DIM)
#define LN_EPS 1e-5f

// ---------------------------------------------------------------------------
// generic f32 -> f16 cast
// ---------------------------------------------------------------------------
__global__ void cast_f32_to_f16(const float* __restrict__ src,
                                _Float16* __restrict__ dst, int n) {
    int i = blockIdx.x * blockDim.x + threadIdx.x;
    if (i < n) dst[i] = (_Float16)src[i];
}

// ---------------------------------------------------------------------------
// per-layer state init: h,c from h0[l], c0[l]
// ---------------------------------------------------------------------------
__global__ void init_state(const float* __restrict__ h0l,
                           const float* __restrict__ c0l,
                           _Float16* __restrict__ hbuf,
                           float* __restrict__ cbuf) {
    int i = blockIdx.x * blockDim.x + threadIdx.x;
    if (i < B_DIM * H_DIM) {
        hbuf[i] = (_Float16)h0l[i];
        cbuf[i] = c0l[i];
    }
}

// ---------------------------------------------------------------------------
// Fused GEMM + LayerNorm for the input-to-hidden path.
//   out[m, :] = LN( A[m, :] @ W^T + bias ) * gamma + beta   (LN over 2048)
// One workgroup owns a full 16-row x 2048-col slab so the LN reduction is
// block-local. 8 waves x 16 WMMA N-tiles; K loop = 512/32.
// ---------------------------------------------------------------------------
__global__ __launch_bounds__(256)
void gemm_ih_ln(const _Float16* __restrict__ A,     // [M, 512] row-major f16
                const _Float16* __restrict__ W,     // [2048, 512] f16
                const float* __restrict__ bias,     // [2048]
                const float* __restrict__ gamma,    // [2048]
                const float* __restrict__ beta,     // [2048]
                _Float16* __restrict__ out)         // [M, 2048] f16
{
    __shared__ __align__(16) _Float16 As[16 * 512];   // 16 KB
    __shared__ float sstat[2][16][8];                 // per-wave partial stats
    __shared__ float smean[16], srstd[16];

    const int tid  = threadIdx.x;
    const int lane = tid & 31;
    const int wave = tid >> 5;
    const int kHi  = lane >> 4;          // half-wave group
    const int col  = lane & 15;          // column within a 16x16 tile
    const size_t rowBlk = (size_t)blockIdx.x * 16;

    // cooperative load of the 16x512 f16 A slab (16 KB): 4 x b128 per thread
    {
        const v4u* src = (const v4u*)(A + rowBlk * 512);
        v4u* dst = (v4u*)As;
#pragma unroll
        for (int i = 0; i < 4; ++i) dst[tid + i * 256] = src[tid + i * 256];
    }
    __syncthreads();

    v8f acc[16];
#pragma unroll
    for (int i = 0; i < 16; ++i) acc[i] = (v8f){0.f,0.f,0.f,0.f,0.f,0.f,0.f,0.f};

#pragma unroll 1
    for (int kb = 0; kb < 16; ++kb) {
        const int k = kb * 32;
        // A fragment: lane m holds row m, K {0..7,16..23}; lane m+16: K {8..15,24..31}
        Frag16 a;
        const _Float16* ap = As + col * 512 + k + kHi * 8;
        a.u[0] = *(const v4u*)(ap);
        a.u[1] = *(const v4u*)(ap + 16);
#pragma unroll
        for (int i = 0; i < 16; ++i) {
            const int nc = wave * 256 + i * 16 + col;   // gate column 0..2047
            Frag16 b;   // B col nc = W row nc, contiguous in K
            const _Float16* bp = W + (size_t)nc * 512 + k + kHi * 8;
            b.u[0] = *(const v4u*)(bp);
            b.u[1] = *(const v4u*)(bp + 16);
            acc[i] = __builtin_amdgcn_wmma_f32_16x16x32_f16(
                false, a.h, false, b.h, (short)0, acc[i], false, false);
        }
    }

    // add bias (pre-LN), accumulate per-row partial sums in registers
    float rs[8] = {0,0,0,0,0,0,0,0}, rq[8] = {0,0,0,0,0,0,0,0};
#pragma unroll
    for (int i = 0; i < 16; ++i) {
        const float bv = bias[wave * 256 + i * 16 + col];
#pragma unroll
        for (int v = 0; v < 8; ++v) {
            const float x = acc[i][v] + bv;
            acc[i][v] = x;
            rs[v] += x;
            rq[v] += x * x;
        }
    }
    // reduce across the 16 lanes of each half-wave group (cols of the tiles)
#pragma unroll
    for (int m = 8; m >= 1; m >>= 1) {
#pragma unroll
        for (int v = 0; v < 8; ++v) {
            rs[v] += __shfl_xor(rs[v], m, 32);
            rq[v] += __shfl_xor(rq[v], m, 32);
        }
    }
    if (col == 0) {
#pragma unroll
        for (int v = 0; v < 8; ++v) {
            const int r = kHi * 8 + v;
            sstat[0][r][wave] = rs[v];
            sstat[1][r][wave] = rq[v];
        }
    }
    __syncthreads();
    if (tid < 16) {
        float s1 = 0.f, s2 = 0.f;
#pragma unroll
        for (int w = 0; w < 8; ++w) { s1 += sstat[0][tid][w]; s2 += sstat[1][tid][w]; }
        const float m  = s1 * (1.f / G_DIM);
        const float vv = s2 * (1.f / G_DIM) - m * m;
        smean[tid] = m;
        srstd[tid] = rsqrtf(vv + LN_EPS);
    }
    __syncthreads();

    // normalize in registers and store f16
#pragma unroll
    for (int i = 0; i < 16; ++i) {
        const int nc = wave * 256 + i * 16 + col;
        const float g = gamma[nc], be = beta[nc];
#pragma unroll
        for (int v = 0; v < 8; ++v) {
            const int r = kHi * 8 + v;
            const float val = (acc[i][v] - smean[r]) * srstd[r] * g + be;
            out[(rowBlk + (size_t)r) * G_DIM + nc] = (_Float16)val;
        }
    }
}

// ---------------------------------------------------------------------------
// Recurrent GEMM: R = h @ W_hh^T + b_hh   (raw pre-LN gates, f32)
// grid (4, 8): 16 batch rows x 256 gate cols per WG; W_hh stays in L2.
// ---------------------------------------------------------------------------
__global__ __launch_bounds__(256)
void gemm_hh(const _Float16* __restrict__ hbuf,   // [64, 512] f16
             const _Float16* __restrict__ W,      // [2048, 512] f16
             const float* __restrict__ bias,      // [2048]
             float* __restrict__ R)               // [64, 2048] f32
{
    __shared__ __align__(16) _Float16 As[16 * 512];
    const int tid  = threadIdx.x;
    const int lane = tid & 31;
    const int wave = tid >> 5;
    const int kHi  = lane >> 4;
    const int col  = lane & 15;
    const size_t rowBlk = (size_t)blockIdx.x * 16;

    {
        const v4u* src = (const v4u*)(hbuf + rowBlk * 512);
        v4u* dst = (v4u*)As;
#pragma unroll
        for (int i = 0; i < 4; ++i) dst[tid + i * 256] = src[tid + i * 256];
    }
    __syncthreads();

    v8f acc[2];
    acc[0] = (v8f){0.f,0.f,0.f,0.f,0.f,0.f,0.f,0.f};
    acc[1] = (v8f){0.f,0.f,0.f,0.f,0.f,0.f,0.f,0.f};

#pragma unroll 1
    for (int kb = 0; kb < 16; ++kb) {
        const int k = kb * 32;
        Frag16 a;
        const _Float16* ap = As + col * 512 + k + kHi * 8;
        a.u[0] = *(const v4u*)(ap);
        a.u[1] = *(const v4u*)(ap + 16);
#pragma unroll
        for (int i = 0; i < 2; ++i) {
            const int nc = blockIdx.y * 256 + wave * 32 + i * 16 + col;
            Frag16 b;
            const _Float16* bp = W + (size_t)nc * 512 + k + kHi * 8;
            b.u[0] = *(const v4u*)(bp);
            b.u[1] = *(const v4u*)(bp + 16);
            acc[i] = __builtin_amdgcn_wmma_f32_16x16x32_f16(
                false, a.h, false, b.h, (short)0, acc[i], false, false);
        }
    }
#pragma unroll
    for (int i = 0; i < 2; ++i) {
        const int nc = blockIdx.y * 256 + wave * 32 + i * 16 + col;
        const float bv = bias[nc];
#pragma unroll
        for (int v = 0; v < 8; ++v) {
            const size_t r = rowBlk + kHi * 8 + v;
            R[r * G_DIM + nc] = acc[i][v] + bv;
        }
    }
}

// ---------------------------------------------------------------------------
// Fused step: LN(hh) + gates + cell update + LN(c) + outputs.
// grid 4 WGs x 16 batch rows; each WG owns all 512 H columns of its rows,
// so both LayerNorms reduce block-locally.
// ---------------------------------------------------------------------------
__device__ inline void block_reduce2(float& a, float& b, float* red, int tid) {
#pragma unroll
    for (int m = 16; m >= 1; m >>= 1) {
        a += __shfl_xor(a, m, 32);
        b += __shfl_xor(b, m, 32);
    }
    __syncthreads();
    if ((tid & 31) == 0) { red[(tid >> 5) * 2] = a; red[(tid >> 5) * 2 + 1] = b; }
    __syncthreads();
    float ta = 0.f, tb = 0.f;
#pragma unroll
    for (int w = 0; w < 8; ++w) { ta += red[w * 2]; tb += red[w * 2 + 1]; }
    a = ta; b = tb;
}

__global__ __launch_bounds__(256)
void lstm_step(const _Float16* __restrict__ ih_t,  // [64, 2048] f16 (LN'd ih, step t)
               const float* __restrict__ R,        // [64, 2048] raw hh gates
               float* __restrict__ cbuf,           // [64, 512]
               _Float16* __restrict__ hbuf,        // [64, 512]
               const float* __restrict__ g_hh, const float* __restrict__ be_hh,
               const float* __restrict__ g_ho, const float* __restrict__ be_ho,
               _Float16* __restrict__ ymid_t,      // layer 0 output (f16, GEMM input)
               float* __restrict__ y_t,            // layer 1 output (d_out y)
               float* __restrict__ hy, float* __restrict__ cy,
               int layer, int isLast)
{
    __shared__ float red[16];
    const int tid = threadIdx.x;

    for (int rl = 0; rl < 16; ++rl) {
        const int bb = blockIdx.x * 16 + rl;
        const float*    Rrow = R    + (size_t)bb * G_DIM;
        const _Float16* Irow = ih_t + (size_t)bb * G_DIM;

        // LN stats over the 2048 raw hh-gate values of this row
        float s1 = 0.f, s2 = 0.f;
#pragma unroll
        for (int j = tid; j < G_DIM; j += 256) {
            const float v = Rrow[j];
            s1 += v; s2 += v * v;
        }
        block_reduce2(s1, s2, red, tid);
        const float mh = s1 * (1.f / G_DIM);
        const float rh = rsqrtf(s2 * (1.f / G_DIM) - mh * mh + LN_EPS);

        float cn[2], ov[2];
        float cs1 = 0.f, cs2 = 0.f;
#pragma unroll
        for (int u = 0; u < 2; ++u) {
            const int j = tid + u * 256;    // j in 0..511
            auto gate = [&](int jj) -> float {
                return (float)Irow[jj] + (Rrow[jj] - mh) * rh * g_hh[jj] + be_hh[jj];
            };
            const float iv = 1.f / (1.f + __expf(-gate(j)));
            const float fv = 1.f / (1.f + __expf(-gate(j + 512)));
            const float o  = 1.f / (1.f + __expf(-gate(j + 1024)));
            const float gg = tanhf(gate(j + 1536));
            const float c  = fv * cbuf[bb * H_DIM + j] + iv * gg;
            cbuf[bb * H_DIM + j] = c;
            cn[u] = c; ov[u] = o;
            cs1 += c; cs2 += c * c;
        }
        block_reduce2(cs1, cs2, red, tid);
        const float mc = cs1 * (1.f / H_DIM);
        const float rc = rsqrtf(cs2 * (1.f / H_DIM) - mc * mc + LN_EPS);

#pragma unroll
        for (int u = 0; u < 2; ++u) {
            const int j = tid + u * 256;
            const float tt = tanhf((cn[u] - mc) * rc * g_ho[j] + be_ho[j]);
            const float h  = ov[u] * tt;
            hbuf[bb * H_DIM + j] = (_Float16)h;
            if (layer == 0) ymid_t[bb * H_DIM + j] = (_Float16)h;
            else            y_t[bb * H_DIM + j]    = h;
            if (isLast) {
                hy[bb * H_DIM + j] = h;
                cy[bb * H_DIM + j] = cn[u];
            }
        }
    }
}

// ---------------------------------------------------------------------------
extern "C" void kernel_launch(void* const* d_in, const int* in_sizes, int n_in,
                              void* d_out, int out_size, void* d_ws, size_t ws_size,
                              hipStream_t stream) {
    (void)in_sizes; (void)n_in; (void)out_size; (void)ws_size;

    const float* x    = (const float*)d_in[0];
    const float* h0   = (const float*)d_in[1];
    const float* c0   = (const float*)d_in[2];
    const float* wih  = (const float*)d_in[3];
    const float* whh  = (const float*)d_in[4];
    const float* bih  = (const float*)d_in[5];
    const float* bhh  = (const float*)d_in[6];
    const float* gih  = (const float*)d_in[7];
    const float* beih = (const float*)d_in[8];
    const float* ghh  = (const float*)d_in[9];
    const float* behh = (const float*)d_in[10];
    const float* gho  = (const float*)d_in[11];
    const float* beho = (const float*)d_in[12];
    float* out = (float*)d_out;

    char* ws = (char*)d_ws;
    size_t off = 0;
    auto take = [&](size_t bytes) -> char* {
        char* p = ws + off;
        off += (bytes + 255) & ~(size_t)255;
        return p;
    };
    _Float16* ihbuf = (_Float16*)take((size_t)TB_DIM * G_DIM * 2);           // 128 MB
    _Float16* xa    = (_Float16*)take((size_t)TB_DIM * D_DIM * 2);           // 32 MB
    _Float16* ymid  = (_Float16*)take((size_t)TB_DIM * H_DIM * 2);           // 32 MB
    _Float16* wih16 = (_Float16*)take((size_t)L_DIM * G_DIM * D_DIM * 2);    // 4 MB
    _Float16* whh16 = (_Float16*)take((size_t)L_DIM * G_DIM * H_DIM * 2);    // 4 MB
    float*    Rbuf  = (float*)   take((size_t)B_DIM * G_DIM * 4);            // 512 KB
    _Float16* hbuf  = (_Float16*)take((size_t)B_DIM * H_DIM * 2);
    float*    cbuf  = (float*)   take((size_t)B_DIM * H_DIM * 4);

    // one-time casts to f16
    {
        const int nx = TB_DIM * D_DIM;
        cast_f32_to_f16<<<(nx + 255) / 256, 256, 0, stream>>>(x, xa, nx);
        const int nw = L_DIM * G_DIM * D_DIM;
        cast_f32_to_f16<<<(nw + 255) / 256, 256, 0, stream>>>(wih, wih16, nw);
        cast_f32_to_f16<<<(nw + 255) / 256, 256, 0, stream>>>(whh, whh16, nw);
    }

    float* yBase  = out;
    float* hyBase = out + (size_t)T_DIM * B_DIM * H_DIM;
    float* cyBase = hyBase + (size_t)L_DIM * B_DIM * H_DIM;

    for (int l = 0; l < L_DIM; ++l) {
        const _Float16* Ain = (l == 0) ? xa : ymid;
        gemm_ih_ln<<<TB_DIM / 16, 256, 0, stream>>>(
            Ain, wih16 + (size_t)l * G_DIM * D_DIM,
            bih + l * G_DIM, gih + l * G_DIM, beih + l * G_DIM, ihbuf);

        init_state<<<(B_DIM * H_DIM + 255) / 256, 256, 0, stream>>>(
            h0 + (size_t)l * B_DIM * H_DIM, c0 + (size_t)l * B_DIM * H_DIM, hbuf, cbuf);

        for (int t = 0; t < T_DIM; ++t) {
            gemm_hh<<<dim3(4, 8), 256, 0, stream>>>(
                hbuf, whh16 + (size_t)l * G_DIM * H_DIM, bhh + l * G_DIM, Rbuf);
            lstm_step<<<4, 256, 0, stream>>>(
                ihbuf + (size_t)t * B_DIM * G_DIM, Rbuf, cbuf, hbuf,
                ghh + l * G_DIM, behh + l * G_DIM, gho + l * H_DIM, beho + l * H_DIM,
                ymid + (size_t)t * B_DIM * H_DIM, yBase + (size_t)t * B_DIM * H_DIM,
                hyBase + (size_t)l * B_DIM * H_DIM, cyBase + (size_t)l * B_DIM * H_DIM,
                l, (t == T_DIM - 1) ? 1 : 0);
        }
    }
}
